// CooperativePolicy_61409442398975
// MI455X (gfx1250) — matrix operand order repaired and checked by reading, not compiled, not run on hardware
//
#include <hip/hip_runtime.h>
#include <hip/hip_bf16.h>
#include <math.h>

typedef __attribute__((ext_vector_type(16))) _Float16 v16h;
typedef __attribute__((ext_vector_type(8)))  _Float16 v8h;
typedef __attribute__((ext_vector_type(8)))  float    v8f;

// ---------------------------------------------------------------------------
// WMMA fragment I/O per CDNA5 ISA 7.12.2 (wave32).
// A: 16x32 f16. lane<16: row=lane, K {0..7,16..23}; lane>=16: same row,
// K {8..15,24..31}. Both halves are 16B-contiguous runs -> ds_load_b128 x2.
// ---------------------------------------------------------------------------
__device__ __forceinline__ v16h load_A(const _Float16* __restrict__ src, int lda, int lane) {
    int row = lane & 15;
    int kb  = (lane & 16) ? 8 : 0;
    const _Float16* p = src + row * lda + kb;
    v8h lo = *(const v8h*)(p);
    v8h hi = *(const v8h*)(p + 16);
    return __builtin_shufflevector(lo, hi, 0,1,2,3,4,5,6,7,8,9,10,11,12,13,14,15);
}

// B fragments are pre-swizzled: 32 contiguous bytes per lane -> global b128 x2.
__device__ __forceinline__ v16h load_B_swz(const _Float16* __restrict__ frag, int lane) {
    const _Float16* p = frag + lane * 16;
    v8h lo = *(const v8h*)(p);
    v8h hi = *(const v8h*)(p + 8);
    return __builtin_shufflevector(lo, hi, 0,1,2,3,4,5,6,7,8,9,10,11,12,13,14,15);
}

// Warm WGP$/L2 with a weight panel (emits global_prefetch_b8).
__device__ __forceinline__ void prefetch_panel(const void* p, int bytes, int tid) {
    for (int off = tid * 256; off < bytes; off += 256 * 256)
        __builtin_prefetch((const char*)p + off, 0, 1);
}

// ---------------------------------------------------------------------------
// Block GEMM: 8 waves; each wave owns 16x64 output strips (4 accumulators),
// one A fragment feeds 4 WMMAs per k-chunk. W is the swizzled f16 image:
// fragment f = kc*COLT + ct occupies 512 f16 starting at f*512.
// ---------------------------------------------------------------------------
template<int ROWT, int COLT, int KCH, bool RELU, typename OutT>
__device__ __forceinline__ void gemm_block(
    const _Float16* __restrict__ A, int lda,
    const _Float16* __restrict__ W,
    const float* __restrict__ bias,
    OutT* __restrict__ C, int ldc, int tid)
{
    const int wave = tid >> 5, lane = tid & 31;
    const int col = lane & 15;
    const int rbase = (lane & 16) ? 8 : 0;
    constexpr int CGRP = COLT / 4;
    constexpr int GROUPS = ROWT * CGRP;
    for (int g = wave; g < GROUPS; g += 8) {
        int rt = g % ROWT, ct0 = (g / ROWT) * 4;
        v8f acc[4];
#pragma unroll
        for (int c = 0; c < 4; ++c) {
            float bv = bias[(ct0 + c) * 16 + col];
#pragma unroll
            for (int i = 0; i < 8; ++i) acc[c][i] = bv;
        }
#pragma unroll
        for (int kc = 0; kc < KCH; ++kc) {
            v16h a = load_A(A + (rt * 16) * lda + kc * 32, lda, lane);
#pragma unroll
            for (int c = 0; c < 4; ++c) {
                v16h b = load_B_swz(W + (size_t)(kc * COLT + ct0 + c) * 512, lane);
                acc[c] = __builtin_amdgcn_wmma_f32_16x16x32_f16(false, a, false, b,
                                                                (short)0, acc[c], false, false);
            }
        }
#pragma unroll
        for (int c = 0; c < 4; ++c) {
#pragma unroll
            for (int r = 0; r < 8; ++r) {
                float v = acc[c][r];
                if (RELU && v < 0.f) v = 0.f;
                C[(rt * 16 + rbase + r) * ldc + (ct0 + c) * 16 + col] = (OutT)v;
            }
        }
    }
}

// ---------------------------------------------------------------------------
// Kernel 0: f32 weights -> f16 image in WMMA B-fragment order, zero padding.
// dst[f*512 + l*16 + j] = W[kc*32 + (l&16)/2 + (j<8 ? j : j+8)][ct*16 + (l&15)], f=kc*colT+ct
// ---------------------------------------------------------------------------
__global__ void cvt_swz_kernel(const float* __restrict__ src, _Float16* __restrict__ dst,
                               int rows, int cols, int padRows) {
    int o = blockIdx.x * blockDim.x + threadIdx.x;
    int total = padRows * cols;
    if (o >= total) return;
    int j = o & 15;
    int l = (o >> 4) & 31;
    int f = o >> 9;
    int colTiles = cols >> 4;
    int ct = f % colTiles;
    int kc = f / colTiles;
    int k = kc * 32 + ((l & 16) ? 8 : 0) + (j < 8 ? j : j + 8);
    int c = ct * 16 + (l & 15);
    dst[o] = (k < rows) ? (_Float16)src[k * cols + c] : (_Float16)0.f;
}

// ---------------------------------------------------------------------------
// Kernel 1: LayerNorm -> enc1 -> enc2 -> intent -> L2 normalize. 64 nodes/blk.
// ---------------------------------------------------------------------------
__global__ __launch_bounds__(256) void encoder_kernel(
    const float* __restrict__ x, const float* __restrict__ lg, const float* __restrict__ lb,
    const _Float16* __restrict__ w1, const float* __restrict__ b1,
    const _Float16* __restrict__ w2, const float* __restrict__ b2,
    const _Float16* __restrict__ wi, const float* __restrict__ bi,
    float* __restrict__ zf, _Float16* __restrict__ zh, int Nn)
{
    __shared__ __align__(16) char smem[65536];
    _Float16* A0 = (_Float16*)smem;              // 64x128 f16 (16KB)
    _Float16* H1 = (_Float16*)(smem + 32768);    // 64x256 f16 (32KB)
    _Float16* H2 = (_Float16*)smem;              // 64x256 f16 (reuses A0 region)
    float*    Zl = (float*)(smem + 32768);       // 64x64  f32 (reuses H1 region)

    const int row0 = blockIdx.x * 64;
    const int tid = threadIdx.x;

    prefetch_panel(w1, 128 * 256 * 2, tid);
    prefetch_panel(w2, 256 * 256 * 2, tid);

    if (tid < 64) {                              // LayerNorm, one row per thread
        int gr = row0 + tid;
        _Float16* ar = A0 + tid * 128;
        if (gr < Nn) {
            const float* xp = x + (size_t)gr * 128;
            float mu = 0.f;
            for (int k = 0; k < 128; ++k) mu += xp[k];
            mu *= (1.f / 128.f);
            float var = 0.f;
            for (int k = 0; k < 128; ++k) { float d = xp[k] - mu; var += d * d; }
            var *= (1.f / 128.f);
            float inv = rsqrtf(var + 1e-5f);
            for (int k = 0; k < 128; ++k)
                ar[k] = (_Float16)((xp[k] - mu) * inv * lg[k] + lb[k]);
        } else {
            for (int k = 0; k < 128; ++k) ar[k] = (_Float16)0.f;
        }
    }
    __syncthreads();
    gemm_block<4, 16, 4, true,  _Float16>(A0, 128, w1, b1, H1, 256, tid);
    __syncthreads();
    gemm_block<4, 16, 8, true,  _Float16>(H1, 256, w2, b2, H2, 256, tid);
    __syncthreads();
    gemm_block<4,  4, 8, false, float   >(H2, 256, wi, bi, Zl,  64, tid);
    __syncthreads();
    if (tid < 64) {                              // F.normalize
        int gr = row0 + tid;
        if (gr < Nn) {
            const float* zr = Zl + tid * 64;
            float ss = 0.f;
            for (int k = 0; k < 64; ++k) ss += zr[k] * zr[k];
            float s = 1.f / fmaxf(sqrtf(ss), 1e-12f);
            for (int k = 0; k < 64; ++k) {
                float v = zr[k] * s;
                zf[(size_t)gr * 64 + k] = v;
                zh[(size_t)gr * 64 + k] = (_Float16)v;
            }
        }
    }
}

// ---------------------------------------------------------------------------
// Kernel 2: per-node GAT attention. K/V via WMMA, softmax, wo -> feat[N][128].
// ---------------------------------------------------------------------------
__global__ __launch_bounds__(256) void attn_kernel(
    const float* __restrict__ rel_pos, const int* __restrict__ nbr_idx,
    const int* __restrict__ nbr_mask,
    const float* __restrict__ zf, const _Float16* __restrict__ zh,
    const float* __restrict__ wq_w, const float* __restrict__ wq_b,
    const _Float16* __restrict__ wk_h, const float* __restrict__ wk_b,
    const _Float16* __restrict__ wv_h, const float* __restrict__ wv_b,
    const float* __restrict__ wo_w, const float* __restrict__ wo_b,
    _Float16* __restrict__ feat)
{
    __shared__ _Float16 nfh[32 * 96];
    __shared__ _Float16 Kh[32 * 256];
    __shared__ _Float16 Vh[32 * 256];
    __shared__ float Qs[256];
    __shared__ float sc[32];
    __shared__ float ew[32];
    __shared__ float wt[32];
    __shared__ float ctx[256];
    __shared__ int idxs[32];
    __shared__ int msks[32];

    const int n = blockIdx.x;
    const int tid = threadIdx.x;
    const int lane = tid & 31;

    if (tid < 32) {
        int ii = nbr_idx[n * 32 + tid];
        idxs[tid] = ii < 0 ? 0 : ii;
        msks[tid] = nbr_mask[n * 32 + tid];
    }
    __syncthreads();

    for (int e = tid; e < 32 * 96; e += 256) {   // build nf (gathers hit L2)
        int m = e / 96, c = e - m * 96;
        _Float16 v;
        if (c < 64)      v = zh[(size_t)idxs[m] * 64 + c];
        else if (c < 68) v = (_Float16)rel_pos[((size_t)n * 32 + m) * 4 + (c - 64)];
        else             v = (_Float16)0.f;
        nfh[e] = v;
    }
    {                                            // Q = wq(z), one col per thread
        float q = wq_b[tid];
        const float* zp = zf + (size_t)n * 64;
        for (int k = 0; k < 64; ++k) q += zp[k] * wq_w[k * 256 + tid];
        Qs[tid] = q;
    }
    __syncthreads();

    {   // K (waves 0-3) / V (waves 4-7): 2 row tiles x 16 col tiles, K=96
        int wave = tid >> 5;
        const _Float16* Wg = (wave < 4) ? wk_h : wv_h;
        const float*    bb = (wave < 4) ? wk_b : wv_b;
        _Float16*       Cl = (wave < 4) ? Kh : Vh;
        int col = lane & 15;
        int rbase = (lane & 16) ? 8 : 0;
        for (int g = (wave & 3); g < 8; g += 4) {   // 2 rowT x 4 col groups
            int rt = g & 1, ct0 = (g >> 1) * 4;
            v8f acc[4];
#pragma unroll
            for (int c = 0; c < 4; ++c) {
                float bv = bb[(ct0 + c) * 16 + col];
#pragma unroll
                for (int i = 0; i < 8; ++i) acc[c][i] = bv;
            }
#pragma unroll
            for (int kc = 0; kc < 3; ++kc) {
                v16h a = load_A(nfh + rt * 16 * 96 + kc * 32, 96, lane);
#pragma unroll
                for (int c = 0; c < 4; ++c) {
                    v16h b = load_B_swz(Wg + (size_t)(kc * 16 + ct0 + c) * 512, lane);
                    acc[c] = __builtin_amdgcn_wmma_f32_16x16x32_f16(false, a, false, b,
                                                                    (short)0, acc[c], false, false);
                }
            }
#pragma unroll
            for (int c = 0; c < 4; ++c)
#pragma unroll
                for (int r = 0; r < 8; ++r)
                    Cl[(rt * 16 + rbase + r) * 256 + (ct0 + c) * 16 + col] = (_Float16)acc[c][r];
        }
    }
    __syncthreads();

    {                                            // scores: 8 lanes per neighbor
        int m = tid >> 3, sl = tid & 7;
        float s = 0.f;
        const _Float16* kp = Kh + m * 256;
        for (int h = sl; h < 256; h += 8) s += Qs[h] * (float)kp[h];
        s += __shfl_xor(s, 1, 32);
        s += __shfl_xor(s, 2, 32);
        s += __shfl_xor(s, 4, 32);
        if (sl == 0) {
            s *= 0.0625f;                        // 1/sqrt(256)
            if (msks[m] == 0) s = -1e9f;
            sc[m] = s;
        }
    }
    __syncthreads();
    if (tid < 32) {
        float mx = sc[0];
        for (int j = 1; j < 32; ++j) mx = fmaxf(mx, sc[j]);
        ew[tid] = expf(sc[tid] - mx);
    }
    __syncthreads();
    if (tid < 32) {
        float sum = 0.f;
        for (int j = 0; j < 32; ++j) sum += ew[j];
        wt[tid] = ew[tid] / sum;
    }
    __syncthreads();
    {                                            // ctx = sum_m w[m] * V[m]
        float c = 0.f;
        for (int m = 0; m < 32; ++m) c += wt[m] * (float)Vh[m * 256 + tid];
        ctx[tid] = c;
    }
    __syncthreads();
    {                                            // context = wo(ctx): 4 lanes/out
        int j = tid >> 2, sl = tid & 3;
        float o = 0.f;
        for (int h = sl; h < 256; h += 4) o += ctx[h] * wo_w[h * 64 + j];
        o += __shfl_xor(o, 1, 32);
        o += __shfl_xor(o, 2, 32);
        if (sl == 0) {
            o += wo_b[j];
            feat[(size_t)n * 128 + 64 + j] = (_Float16)o;
            feat[(size_t)n * 128 + j]      = (_Float16)zf[(size_t)n * 64 + j];
        }
    }
}

// ---------------------------------------------------------------------------
// Kernel 3: actor (act1,act2,mean,std) + critic (cr1,cr2,cr3). 64 nodes/blk.
// ---------------------------------------------------------------------------
__global__ __launch_bounds__(256) void heads_kernel(
    const _Float16* __restrict__ feat,
    const _Float16* __restrict__ a1w, const float* __restrict__ a1b,
    const _Float16* __restrict__ a2w, const float* __restrict__ a2b,
    const float* __restrict__ mean_w, const float* __restrict__ mean_b,
    const float* __restrict__ std_w,  const float* __restrict__ std_b,
    const _Float16* __restrict__ c1w, const float* __restrict__ c1b,
    const _Float16* __restrict__ c2w, const float* __restrict__ c2b,
    const float* __restrict__ cr3_w,  const float* __restrict__ cr3_b,
    float* __restrict__ out_mean, float* __restrict__ out_std,
    float* __restrict__ out_val, int Nn)
{
    __shared__ __align__(16) char smem[65536];
    _Float16* R0 = (_Float16*)smem;              // 32KB region
    _Float16* R1 = (_Float16*)(smem + 32768);    // 32KB region

    const int row0 = blockIdx.x * 64;
    const int tid = threadIdx.x;

    prefetch_panel(a2w, 256 * 256 * 2, tid);
    prefetch_panel(c2w, 256 * 256 * 2, tid);

    for (int e = tid; e < 64 * 128; e += 256) {  // stage feat -> R0
        int r = e >> 7, gr = row0 + r;
        R0[e] = (gr < Nn) ? feat[(size_t)gr * 128 + (e & 127)] : (_Float16)0.f;
    }
    __syncthreads();
    gemm_block<4, 16, 4, true, _Float16>(R0, 128, a1w, a1b, R1, 256, tid);
    __syncthreads();
    gemm_block<4, 16, 8, true, _Float16>(R1, 256, a2w, a2b, R0, 256, tid);
    __syncthreads();
    for (int e = tid; e < 64 * 128; e += 256) {  // restage feat -> R1 (critic)
        int r = e >> 7, gr = row0 + r;
        R1[e] = (gr < Nn) ? feat[(size_t)gr * 128 + (e & 127)] : (_Float16)0.f;
    }
    {                                            // actor heads: (node, j) per thread
        int r = tid >> 2, j = tid & 3;
        int gr = row0 + r;
        if (gr < Nn) {
            const _Float16* hp = R0 + r * 256;
            if (j < 2) {
                float v = mean_b[j];
                for (int k = 0; k < 256; ++k) v += (float)hp[k] * mean_w[k * 2 + j];
                out_mean[(size_t)gr * 2 + j] = v;
            } else {
                int jj = j - 2;
                float v = std_b[jj];
                for (int k = 0; k < 256; ++k) v += (float)hp[k] * std_w[k * 2 + jj];
                float sp = (v > 20.f) ? v : log1pf(expf(v));
                out_std[(size_t)gr * 2 + jj] = sp + 1e-5f;
            }
        }
    }
    __syncthreads();
    gemm_block<4, 16, 4, true, _Float16>(R1, 128, c1w, c1b, R0, 256, tid);
    __syncthreads();
    gemm_block<4, 16, 8, true, _Float16>(R0, 256, c2w, c2b, R1, 256, tid);
    __syncthreads();
    if (tid < 64) {
        int gr = row0 + tid;
        if (gr < Nn) {
            float v = cr3_b[0];
            const _Float16* hp = R1 + tid * 256;
            for (int k = 0; k < 256; ++k) v += (float)hp[k] * cr3_w[k];
            out_val[gr] = v;
        }
    }
}

// ---------------------------------------------------------------------------
extern "C" void kernel_launch(void* const* d_in, const int* in_sizes, int n_in,
                              void* d_out, int out_size, void* d_ws, size_t ws_size,
                              hipStream_t stream) {
    if (n_in < 36) return;
    const float* node_features = (const float*)d_in[0];
    const float* rel_pos       = (const float*)d_in[1];
    const int*   nbr_idx       = (const int*)d_in[2];
    const int*   nbr_mask      = (const int*)d_in[3];
    const float* ln_g  = (const float*)d_in[4];
    const float* ln_b  = (const float*)d_in[5];
    const float* enc1_w = (const float*)d_in[6];  const float* enc1_b = (const float*)d_in[7];
    const float* enc2_w = (const float*)d_in[8];  const float* enc2_b = (const float*)d_in[9];
    const float* int_w  = (const float*)d_in[10]; const float* int_b  = (const float*)d_in[11];
    /* aux (12,13) unused by reference */
    const float* wq_w = (const float*)d_in[14]; const float* wq_b = (const float*)d_in[15];
    const float* wk_w = (const float*)d_in[16]; const float* wk_b = (const float*)d_in[17];
    const float* wv_w = (const float*)d_in[18]; const float* wv_b = (const float*)d_in[19];
    const float* wo_w = (const float*)d_in[20]; const float* wo_b = (const float*)d_in[21];
    const float* a1_w = (const float*)d_in[22]; const float* a1_b = (const float*)d_in[23];
    const float* a2_w = (const float*)d_in[24]; const float* a2_b = (const float*)d_in[25];
    const float* mn_w = (const float*)d_in[26]; const float* mn_b = (const float*)d_in[27];
    const float* sd_w = (const float*)d_in[28]; const float* sd_b = (const float*)d_in[29];
    const float* c1_w = (const float*)d_in[30]; const float* c1_b = (const float*)d_in[31];
    const float* c2_w = (const float*)d_in[32]; const float* c2_b = (const float*)d_in[33];
    const float* c3_w = (const float*)d_in[34]; const float* c3_b = (const float*)d_in[35];

    const int N = in_sizes[0] / 128;

    size_t off = 0;
    auto carve = [&](size_t bytes) {
        char* p = (char*)d_ws + off;
        off += (bytes + 255) & ~(size_t)255;
        return p;
    };
    _Float16* h_enc1 = (_Float16*)carve(128 * 256 * 2);
    _Float16* h_enc2 = (_Float16*)carve(256 * 256 * 2);
    _Float16* h_int  = (_Float16*)carve(256 * 64 * 2);
    _Float16* h_wk   = (_Float16*)carve(96 * 256 * 2);   // 68 rows padded to 96
    _Float16* h_wv   = (_Float16*)carve(96 * 256 * 2);
    _Float16* h_a1   = (_Float16*)carve(128 * 256 * 2);
    _Float16* h_a2   = (_Float16*)carve(256 * 256 * 2);
    _Float16* h_c1   = (_Float16*)carve(128 * 256 * 2);
    _Float16* h_c2   = (_Float16*)carve(256 * 256 * 2);
    _Float16* zh     = (_Float16*)carve((size_t)N * 64 * 2);
    float*    zfbuf  = (float*)   carve((size_t)N * 64 * 4);
    _Float16* featbf = (_Float16*)carve((size_t)N * 128 * 2);

    float* out_mean = (float*)d_out;
    float* out_std  = (float*)d_out + (size_t)2 * N;
    float* out_val  = (float*)d_out + (size_t)4 * N;

    auto cvt = [&](const float* s, _Float16* d, int rows, int cols, int padRows) {
        int total = padRows * cols;
        cvt_swz_kernel<<<(total + 255) / 256, 256, 0, stream>>>(s, d, rows, cols, padRows);
    };
    cvt(enc1_w, h_enc1, 128, 256, 128);
    cvt(enc2_w, h_enc2, 256, 256, 256);
    cvt(int_w,  h_int,  256, 64,  256);
    cvt(wk_w,   h_wk,   68,  256, 96);
    cvt(wv_w,   h_wv,   68,  256, 96);
    cvt(a1_w,   h_a1,   128, 256, 128);
    cvt(a2_w,   h_a2,   256, 256, 256);
    cvt(c1_w,   h_c1,   128, 256, 128);
    cvt(c2_w,   h_c2,   256, 256, 256);

    const int nodeBlocks = (N + 63) / 64;
    encoder_kernel<<<nodeBlocks, 256, 0, stream>>>(
        node_features, ln_g, ln_b, h_enc1, enc1_b, h_enc2, enc2_b, h_int, int_b,
        zfbuf, zh, N);
    attn_kernel<<<N, 256, 0, stream>>>(
        rel_pos, nbr_idx, nbr_mask, zfbuf, zh,
        wq_w, wq_b, h_wk, wk_b, h_wv, wv_b, wo_w, wo_b, featbf);
    heads_kernel<<<nodeBlocks, 256, 0, stream>>>(
        featbf, h_a1, a1_b, h_a2, a2_b, mn_w, mn_b, sd_w, sd_b,
        h_c1, c1_b, h_c2, c2_b, c3_w, c3_b, out_mean, out_std, out_val, N);
}
